// BeliefTreeMemory_7181185319307
// MI455X (gfx1250) — matrix operand order repaired and checked by reading, not compiled, not run on hardware
//
#include <hip/hip_runtime.h>
#include <hip/hip_bf16.h>
#include <math.h>

#define N_NODES 200000
#define N_EDGES 400000
#define D_NODE  128
#define D_EDGE  64
#define CAT_DIM 321
#define CAT_PAD 352   // 11 * 32
#define WAVES   8

#define W1_ELEMS  (D_NODE * CAT_PAD)       // 45056
#define W2_ELEMS  (D_NODE * D_NODE)        // 16384
#define WG_ELEMS  (3 * D_NODE * D_NODE)    // 49152 (Wih or Whh)
#define EE_ELEMS  (4 * D_EDGE)             // 256

typedef __attribute__((ext_vector_type(8)))  __bf16 v8bf;
typedef __attribute__((ext_vector_type(16))) __bf16 v16bf;
typedef __attribute__((ext_vector_type(8)))  float  v8f;

__device__ __forceinline__ v16bf ld_frag(const __bf16* p0, const __bf16* p1) {
    v8bf lo = *(const v8bf*)p0;
    v8bf hi = *(const v8bf*)p1;
    return __builtin_shufflevector(lo, hi, 0,1,2,3,4,5,6,7,8,9,10,11,12,13,14,15);
}

__device__ __forceinline__ v8f wmma_bf16(v16bf a, v16bf b, v8f c) {
    return __builtin_amdgcn_wmma_f32_16x16x32_bf16(false, a, false, b, (short)0, c, false, false);
}

__device__ __forceinline__ float sigmoidf_(float x) { return 1.0f / (1.0f + __expf(-x)); }

// intra-wave LDS ordering (per-wave private LDS regions; no block barrier needed)
__device__ __forceinline__ void wave_sync() {
    __builtin_amdgcn_fence(__ATOMIC_RELEASE, "workgroup");
    __builtin_amdgcn_wave_barrier();
    __builtin_amdgcn_fence(__ATOMIC_ACQUIRE, "workgroup");
}

// pack 4 floats -> 4 bf16 -> one 8B LDS store
__device__ __forceinline__ void st4bf(__bf16* p, float4 f) {
    union { __bf16 b[4]; uint2 u; } c;
    c.b[0] = (__bf16)f.x; c.b[1] = (__bf16)f.y;
    c.b[2] = (__bf16)f.z; c.b[3] = (__bf16)f.w;
    *(uint2*)p = c.u;
}

// LDS byte offset of a shared-memory pointer: low 32 bits of the flat address
__device__ __forceinline__ unsigned lds_off(const void* p) {
    return (unsigned)(unsigned long long)(uintptr_t)p;
}

// CDNA5 async DMA: global -> LDS, 16B per lane, ASYNCcnt-tracked
__device__ __forceinline__ void async_ld_b128(unsigned lds, const void* g) {
    asm volatile("global_load_async_to_lds_b128 %0, %1, off"
                 :: "v"(lds), "v"((unsigned long long)(uintptr_t)g)
                 : "memory");
}
__device__ __forceinline__ void wait_async0() {
    asm volatile("s_wait_asynccnt 0x0" ::: "memory");
}

// ---------------------------------------------------------------- utility kernels
__global__ void zero_f32_kernel(float* __restrict__ p, long n4) {
    long i = (long)blockIdx.x * blockDim.x + threadIdx.x;
    if (i < n4) ((float4*)p)[i] = make_float4(0.f, 0.f, 0.f, 0.f);
}

__global__ void count_kernel(const int* __restrict__ tgt, float* __restrict__ cnt) {
    int i = blockIdx.x * blockDim.x + threadIdx.x;
    if (i < N_EDGES) unsafeAtomicAdd(&cnt[tgt[i]], 1.0f);
}

// per-pass: f32 h -> bf16 copy (4 elems per thread)
__global__ void h_to_bf16_kernel(const float* __restrict__ h, __bf16* __restrict__ hb) {
    long i = (long)blockIdx.x * blockDim.x + threadIdx.x;
    if (i < (long)N_NODES * D_NODE / 4) {
        float4 f = ((const float4*)h)[i];
        union { __bf16 b[4]; uint2 u; } c;
        c.b[0] = (__bf16)f.x; c.b[1] = (__bf16)f.y;
        c.b[2] = (__bf16)f.z; c.b[3] = (__bf16)f.w;
        ((uint2*)hb)[i] = c.u;
    }
}

// one-shot f32 -> bf16 weight/embedding conversion (W1 zero-padded to CAT_PAD)
__global__ void cvt_weights_kernel(const float* __restrict__ W1, const float* __restrict__ W2,
                                   const float* __restrict__ Wih, const float* __restrict__ Whh,
                                   const float* __restrict__ eemb,
                                   __bf16* __restrict__ W1b, __bf16* __restrict__ W2b,
                                   __bf16* __restrict__ Wihb, __bf16* __restrict__ Whhb,
                                   __bf16* __restrict__ eembB) {
    int i = blockIdx.x * blockDim.x + threadIdx.x;
    if (i < W1_ELEMS) {
        int n = i / CAT_PAD, k = i - n * CAT_PAD;
        W1b[i] = (k < CAT_DIM) ? (__bf16)W1[n * CAT_DIM + k] : (__bf16)0.0f;
    }
    if (i < W2_ELEMS) W2b[i]  = (__bf16)W2[i];
    if (i < WG_ELEMS) { Wihb[i] = (__bf16)Wih[i]; Whhb[i] = (__bf16)Whh[i]; }
    if (i < EE_ELEMS) eembB[i] = (__bf16)eemb[i];
}

// ---------------------------------------------------------------- edge MLP
__global__ __launch_bounds__(WAVES * 32)
void edge_mlp_kernel(const __bf16* __restrict__ hb,
                     const float* __restrict__ cred,
                     const __bf16* __restrict__ eembB,
                     const __bf16* __restrict__ W1b, const float* __restrict__ b1,
                     const __bf16* __restrict__ W2b, const float* __restrict__ b2,
                     const int* __restrict__ src, const int* __restrict__ tgt,
                     const int* __restrict__ ety,
                     float* __restrict__ agg) {
    __shared__ __align__(16) __bf16 sW1[W1_ELEMS];                //  90112 B
    __shared__ __align__(16) __bf16 sW2[W2_ELEMS];                //  32768 B
    __shared__ __align__(16) __bf16 sMsg[WAVES][16 * CAT_PAD];    //  90112 B
    __shared__ __align__(16) __bf16 sAct[WAVES][16 * D_NODE];     //  32768 B

    const int tid  = threadIdx.x;
    const int lane = tid & 31;
    const int w    = tid >> 5;
    const int lm   = lane & 15;
    const int hi   = lane >> 4;

    // async-DMA weight staging
    {
        unsigned l1 = lds_off(&sW1[0]);
        unsigned l2 = lds_off(&sW2[0]);
        for (int i = tid; i < W1_ELEMS / 8; i += WAVES * 32)
            async_ld_b128(l1 + i * 16, (const uint4*)W1b + i);
        for (int i = tid; i < W2_ELEMS / 8; i += WAVES * 32)
            async_ld_b128(l2 + i * 16, (const uint4*)W2b + i);
        wait_async0();
    }
    __syncthreads();

    // zero the pad region (elements 320..351 of each row) once; element 320
    // (credence) is rewritten per tile, 321..351 stay zero forever.
    for (int i = lane; i < 16 * 8; i += 32) {
        int m = i >> 3, j = i & 7;
        uint2 z; z.x = 0u; z.y = 0u;
        *(uint2*)&sMsg[w][m * CAT_PAD + 320 + j * 4] = z;
    }

    // hoist per-lane bias vectors (invariant across tiles)
    float b1v[8], b2v[8];
    #pragma unroll
    for (int nb = 0; nb < 8; ++nb) { b1v[nb] = b1[nb * 16 + lm]; b2v[nb] = b2[nb * 16 + lm]; }

    const unsigned msgbase = lds_off(&sMsg[w][0]);
    const int NT     = N_EDGES / 16;              // 25000
    const int stride = gridDim.x * WAVES;

    for (int tile = blockIdx.x * WAVES + w; tile < NT; tile += stride) {
        const int e0 = tile * 16;

        // h_src | h_tgt rows: ONE async DMA per row (lanes 0-15 src, 16-31 tgt)
        for (int m = 0; m < 16; ++m) {
            int e = e0 + m;
            long s = src[e], t = tgt[e];
            const __bf16* g = (lane < 16) ? hb + s * D_NODE + lm * 8
                                          : hb + t * D_NODE + lm * 8;
            unsigned loff = msgbase + (unsigned)(m * CAT_PAD * 2)
                          + (lane < 16 ? 0u : 256u) + (unsigned)lm * 16u;
            async_ld_b128(loff, g);
        }
        // edge-type embeddings: 8 chunks of 16B per row
        for (int i = lane; i < 16 * 8; i += 32) {
            int m = i >> 3, j = i & 7;
            int et = ety[e0 + m];
            async_ld_b128(msgbase + (unsigned)(m * CAT_PAD * 2) + 512u + (unsigned)j * 16u,
                          eembB + et * D_EDGE + j * 8);
        }
        // credence element (k = 320)
        if (lane < 16)
            sMsg[w][lane * CAT_PAD + 320] = (__bf16)cred[src[e0 + lane]];
        // prefetch next tile's h rows
        {
            int tnext = tile + stride;
            if (tnext < NT && lane < 16) {
                int e2 = tnext * 16 + lane;
                __builtin_prefetch(&hb[(long)src[e2] * D_NODE], 0, 1);
                __builtin_prefetch(&hb[(long)tgt[e2] * D_NODE], 0, 1);
            }
        }
        wait_async0();
        wave_sync();

        // GEMM1: [16,352] x [352,128]  (batched B loads, then WMMA group)
        v8f acc[8] = {};
        for (int kb = 0; kb < CAT_PAD / 32; ++kb) {
            const __bf16* ap = &sMsg[w][lm * CAT_PAD + kb * 32 + hi * 16];
            v16bf a = ld_frag(ap, ap + 8);
            v16bf bf[8];
            #pragma unroll
            for (int nb = 0; nb < 8; ++nb) {
                const __bf16* bp = &sW1[(nb * 16 + lm) * CAT_PAD + kb * 32 + hi * 16];
                bf[nb] = ld_frag(bp, bp + 8);
            }
            #pragma unroll
            for (int nb = 0; nb < 8; ++nb)
                acc[nb] = wmma_bf16(a, bf[nb], acc[nb]);
        }

        // bias + SiLU -> activation tile
        #pragma unroll
        for (int nb = 0; nb < 8; ++nb) {
            int n = nb * 16 + lm;
            #pragma unroll
            for (int v = 0; v < 8; ++v) {
                float x = acc[nb][v] + b1v[nb];
                sAct[w][(v + hi * 8) * D_NODE + n] = (__bf16)(x * sigmoidf_(x));
            }
        }
        wave_sync();

        // GEMM2: [16,128] x [128,128]
        v8f acc2[8] = {};
        for (int kb = 0; kb < 4; ++kb) {
            const __bf16* ap = &sAct[w][lm * D_NODE + kb * 32 + hi * 16];
            v16bf a = ld_frag(ap, ap + 8);
            v16bf bf[8];
            #pragma unroll
            for (int nb = 0; nb < 8; ++nb) {
                const __bf16* bp = &sW2[(nb * 16 + lm) * D_NODE + kb * 32 + hi * 16];
                bf[nb] = ld_frag(bp, bp + 8);
            }
            #pragma unroll
            for (int nb = 0; nb < 8; ++nb)
                acc2[nb] = wmma_bf16(a, bf[nb], acc2[nb]);
        }

        // scatter-add messages (+b2) into agg[tgt]
        int tv[8];
        #pragma unroll
        for (int v = 0; v < 8; ++v) tv[v] = tgt[e0 + v + hi * 8];
        #pragma unroll
        for (int nb = 0; nb < 8; ++nb) {
            int n = nb * 16 + lm;
            #pragma unroll
            for (int v = 0; v < 8; ++v)
                unsafeAtomicAdd(&agg[(long)tv[v] * D_NODE + n], acc2[nb][v] + b2v[nb]);
        }
        wave_sync();
    }
}

// ---------------------------------------------------------------- GRU cell
__global__ __launch_bounds__(WAVES * 32)
void gru_kernel(const float* __restrict__ agg,
                const float* __restrict__ hprev,
                const __bf16* __restrict__ hb,
                const float* __restrict__ cnt,
                const __bf16* __restrict__ Wihb, const __bf16* __restrict__ Whhb,
                const float* __restrict__ bih, const float* __restrict__ bhh,
                float* __restrict__ dst) {
    __shared__ __align__(16) __bf16 sWih[WG_ELEMS];            // 98304 B
    __shared__ __align__(16) __bf16 sWhh[WG_ELEMS];            // 98304 B
    __shared__ __align__(16) __bf16 sX[WAVES][16 * D_NODE];    // 32768 B
    __shared__ __align__(16) __bf16 sH[WAVES][16 * D_NODE];    // 32768 B

    const int tid  = threadIdx.x;
    const int lane = tid & 31;
    const int w    = tid >> 5;
    const int lm   = lane & 15;
    const int hi   = lane >> 4;

    {
        unsigned li = lds_off(&sWih[0]);
        unsigned lh = lds_off(&sWhh[0]);
        for (int i = tid; i < WG_ELEMS / 8; i += WAVES * 32) {
            async_ld_b128(li + i * 16, (const uint4*)Wihb + i);
            async_ld_b128(lh + i * 16, (const uint4*)Whhb + i);
        }
        wait_async0();
    }
    __syncthreads();

    // hoist per-lane gate biases (invariant across tiles)
    float bR8[8], bZ8[8], bNi8[8], bNh8[8];
    #pragma unroll
    for (int d = 0; d < 8; ++d) {
        int n = d * 16 + lm;
        bR8[d]  = bih[n] + bhh[n];
        bZ8[d]  = bih[128 + n] + bhh[128 + n];
        bNi8[d] = bih[256 + n];
        bNh8[d] = bhh[256 + n];
    }

    const unsigned hbase = lds_off(&sH[w][0]);
    const int NT     = N_NODES / 16;              // 12500
    const int stride = gridDim.x * WAVES;

    for (int tile = blockIdx.x * WAVES + w; tile < NT; tile += stride) {
        const int node0 = tile * 16;

        // h tile via async DMA (2 rows per op: lanes 0-15 row m2, 16-31 row m2+1)
        for (int m2 = 0; m2 < 16; m2 += 2) {
            int m = m2 + hi;
            long node = node0 + m;
            async_ld_b128(hbase + (unsigned)(m * 256) + (unsigned)lm * 16u,
                          hb + node * D_NODE + lm * 8);
        }
        // x = agg/denom (needs f32 scale -> VALU path)
        for (int m = 0; m < 16; ++m) {
            long node = node0 + m;
            float inv = 1.0f / fmaxf(cnt[node], 1.0f);
            float4 xa = ((const float4*)(agg + node * D_NODE))[lane];
            st4bf(&sX[w][m * D_NODE + lane * 4],
                  make_float4(xa.x * inv, xa.y * inv, xa.z * inv, xa.w * inv));
        }
        wait_async0();
        wave_sync();

        // A fragments (K = 128 -> 4 k-blocks each)
        v16bf ax[4], ah[4];
        #pragma unroll
        for (int kb = 0; kb < 4; ++kb) {
            const __bf16* px = &sX[w][lm * D_NODE + kb * 32 + hi * 16];
            const __bf16* ph = &sH[w][lm * D_NODE + kb * 32 + hi * 16];
            ax[kb] = ld_frag(px, px + 8);
            ah[kb] = ld_frag(ph, ph + 8);
        }

        // per 16-col slice: fused gate GEMMs (batched B loads per k-block)
        for (int d = 0; d < 8; ++d) {
            v8f aR = {}, aZ = {}, aNi = {}, aNh = {};
            #pragma unroll
            for (int kb = 0; kb < 4; ++kb) {
                int koff = kb * 32 + hi * 16;
                v16bf bf[6];
                {
                    const __bf16* p0 = &sWih[(      d * 16 + lm) * D_NODE + koff];
                    const __bf16* p1 = &sWhh[(      d * 16 + lm) * D_NODE + koff];
                    const __bf16* p2 = &sWih[(128 + d * 16 + lm) * D_NODE + koff];
                    const __bf16* p3 = &sWhh[(128 + d * 16 + lm) * D_NODE + koff];
                    const __bf16* p4 = &sWih[(256 + d * 16 + lm) * D_NODE + koff];
                    const __bf16* p5 = &sWhh[(256 + d * 16 + lm) * D_NODE + koff];
                    bf[0] = ld_frag(p0, p0 + 8); bf[1] = ld_frag(p1, p1 + 8);
                    bf[2] = ld_frag(p2, p2 + 8); bf[3] = ld_frag(p3, p3 + 8);
                    bf[4] = ld_frag(p4, p4 + 8); bf[5] = ld_frag(p5, p5 + 8);
                }
                aR  = wmma_bf16(ax[kb], bf[0], aR);    // gi_r
                aR  = wmma_bf16(ah[kb], bf[1], aR);    // + gh_r
                aZ  = wmma_bf16(ax[kb], bf[2], aZ);    // gi_z
                aZ  = wmma_bf16(ah[kb], bf[3], aZ);    // + gh_z
                aNi = wmma_bf16(ax[kb], bf[4], aNi);   // gi_n
                aNh = wmma_bf16(ah[kb], bf[5], aNh);   // gh_n
            }
            int n = d * 16 + lm;
            #pragma unroll
            for (int v = 0; v < 8; ++v) {
                long node = node0 + v + hi * 8;
                float r  = sigmoidf_(aR[v] + bR8[d]);
                float z  = sigmoidf_(aZ[v] + bZ8[d]);
                float nn = tanhf((aNi[v] + bNi8[d]) + r * (aNh[v] + bNh8[d]));
                float hv = hprev[node * D_NODE + n];
                float hn = (1.0f - z) * nn + z * hv;
                dst[node * D_NODE + n] = (cnt[node] > 0.0f) ? hn : hv;
            }
        }
        wave_sync();
    }
}

// ---------------------------------------------------------------- launch
extern "C" void kernel_launch(void* const* d_in, const int* in_sizes, int n_in,
                              void* d_out, int out_size, void* d_ws, size_t ws_size,
                              hipStream_t stream) {
    const float* h0   = (const float*)d_in[0];
    const float* cred = (const float*)d_in[1];
    const float* eemb = (const float*)d_in[2];
    const float* W1   = (const float*)d_in[3];
    const float* b1   = (const float*)d_in[4];
    const float* W2   = (const float*)d_in[5];
    const float* b2   = (const float*)d_in[6];
    const float* Wih  = (const float*)d_in[7];
    const float* Whh  = (const float*)d_in[8];
    const float* bih  = (const float*)d_in[9];
    const float* bhh  = (const float*)d_in[10];
    const int*   src  = (const int*)d_in[11];
    const int*   tgt  = (const int*)d_in[12];
    const int*   ety  = (const int*)d_in[13];
    float* out = (float*)d_out;

    char*  ws   = (char*)d_ws;
    float* cnt  = (float*)ws;                                          // 800000 B
    float* bufA = (float*)(ws + (size_t)N_NODES * 4);                  // agg
    float* bufB = bufA + (size_t)N_NODES * D_NODE;                     // h ping-pong
    __bf16* W1b  = (__bf16*)(bufB + (size_t)N_NODES * D_NODE);
    __bf16* W2b  = W1b + W1_ELEMS;
    __bf16* Wihb = W2b + W2_ELEMS;
    __bf16* Whhb = Wihb + WG_ELEMS;
    __bf16* eembB = Whhb + WG_ELEMS;
    __bf16* hb   = eembB + EE_ELEMS;                                   // 51.2 MB bf16 h

    const long nd4  = (long)N_NODES * D_NODE / 4;
    const long cnt4 = N_NODES / 4;

    zero_f32_kernel<<<(cnt4 + 255) / 256, 256, 0, stream>>>(cnt, cnt4);
    count_kernel<<<(N_EDGES + 255) / 256, 256, 0, stream>>>(tgt, cnt);
    cvt_weights_kernel<<<(W1_ELEMS + 255) / 256, 256, 0, stream>>>(
        W1, W2, Wih, Whh, eemb, W1b, W2b, Wihb, Whhb, eembB);

    // pass 0: h0 -> bufB
    h_to_bf16_kernel<<<(nd4 + 255) / 256, 256, 0, stream>>>(h0, hb);
    zero_f32_kernel<<<(nd4 + 255) / 256, 256, 0, stream>>>(bufA, nd4);
    edge_mlp_kernel<<<500, WAVES * 32, 0, stream>>>(hb, cred, eembB, W1b, b1, W2b, b2,
                                                    src, tgt, ety, bufA);
    gru_kernel<<<250, WAVES * 32, 0, stream>>>(bufA, h0, hb, cnt, Wihb, Whhb, bih, bhh, bufB);

    // pass 1: bufB -> d_out
    h_to_bf16_kernel<<<(nd4 + 255) / 256, 256, 0, stream>>>(bufB, hb);
    zero_f32_kernel<<<(nd4 + 255) / 256, 256, 0, stream>>>(bufA, nd4);
    edge_mlp_kernel<<<500, WAVES * 32, 0, stream>>>(hb, cred, eembB, W1b, b1, W2b, b2,
                                                    src, tgt, ety, bufA);
    gru_kernel<<<250, WAVES * 32, 0, stream>>>(bufA, bufB, hb, cnt, Wihb, Whhb, bih, bhh, out);
}
